// MACEModel_57329223467490
// MI455X (gfx1250) — matrix operand order repaired and checked by reading, not compiled, not run on hardware
//
#include <hip/hip_runtime.h>

// ---------------------------------------------------------------------------
// MACE-like equivariant GNN forward, CDNA5 (gfx1250, wave32).
// Dominant FLOPs: per-edge radial-MLP GEMM (8192 x 256 x {6144,11264}) lowered
// to v_wmma_f32_16x16x32_bf16. Everything else is bandwidth/latency bound.
// Scratch (d_ws) usage ~48 MB (W materialized in 512-edge chunks).
// ---------------------------------------------------------------------------

typedef __attribute__((ext_vector_type(16))) __bf16 v16bf;
typedef __attribute__((ext_vector_type(8)))  float  v8f;

constexpr int BS = 128, APM = 32, EMB = 32, EPM = 64;
constexpr int N_NODES = BS * APM;       // 4096
constexpr int N_EDGES = BS * EPM;       // 8192
constexpr int HID  = EMB * 9;           // 288
constexpr int MLP  = 256;
constexpr int NB   = 8;                 // NUM_BESSEL
constexpr int NUM0 = 3 * 2 * EMB * EMB; // 6144  (layer 0, mul_in=64)
constexpr int NUM1 = 11 * EMB * EMB;    // 11264 (layers 1,2)
constexpr int CHUNK = 512;              // edges per materialized-W chunk
constexpr int XSTR  = 3 * APM + EMB;    // 128 floats per x row

// ---- tensor-product path tables (PATHS / COMBOS from the reference) -------
constexpr int NPATH = 11;
constexpr int PL1[NPATH] = {0,0,0,1,1,1,1,2,2,2,2};
constexpr int PL2[NPATH] = {0,1,2,0,1,1,2,0,1,2,2};
constexpr int PL3[NPATH] = {0,1,2,1,0,2,1,2,1,0,2};
constexpr int CGOFF[NPATH] = {0,1,10,35,44,53,98,143,168,213,238};
constexpr int CGTOT = 363;
constexpr int TOFF[NPATH] = {0,1,4,9,12,13,18,21,26,29,30}; // prefix of (2*l3+1)
constexpr int NCOMBO = 41;
constexpr int CBI[NCOMBO] = {0,0,0, 1,1,1,1, 2,2,2,2, 3,3,3,3, 4,4,4,
                             5,5,5,5, 6,6,6,6, 7,7,7,7, 8,8,8,8, 9,9,9, 10,10,10,10};
constexpr int CBJ[NCOMBO] = {0,1,2, 3,4,5,6, 7,8,9,10, 3,4,5,6, 0,1,2,
                             7,8,9,10, 3,4,5,6, 7,8,9,10, 3,4,5,6, 0,1,2, 7,8,9,10};
constexpr int SHOFF[3] = {0,1,4};

__device__ __forceinline__ unsigned short f2bf(float f) { // RNE f32->bf16
  unsigned int u = __float_as_uint(f);
  unsigned int r = u + 0x7FFFu + ((u >> 16) & 1u);
  return (unsigned short)(r >> 16);
}

// ---------------------------------------------------------------------------
// CG coefficients: exact replica of _make_cg (12-pt leggauss x 24 phi, real
// orthonormal SH, per-path Frobenius normalization). One thread per path.
// ---------------------------------------------------------------------------
__device__ void sh_ortho_d(double x, double y, double z, int l, double* out) {
  const double PI = 3.14159265358979323846;
  if (l == 0) { out[0] = 0.5 / sqrt(PI); }
  else if (l == 1) {
    double c = sqrt(3.0 / (4.0 * PI));
    out[0] = c * y; out[1] = c * z; out[2] = c * x;
  } else {
    double a = 0.5 * sqrt(15.0 / PI), b = 0.25 * sqrt(5.0 / PI), c = 0.25 * sqrt(15.0 / PI);
    out[0] = a * x * y; out[1] = a * y * z; out[2] = b * (3.0 * z * z - 1.0);
    out[3] = a * x * z; out[4] = c * (x * x - y * y);
  }
}

__global__ void compute_cg_kernel(float* cg) {
  int p = threadIdx.x;
  if (p >= NPATH) return;
  const double PI = 3.14159265358979323846;
  double ct[12], wt[12];
  for (int i = 0; i < 12; ++i) {          // Newton on P12 -> leggauss(12)
    double x = cos(PI * (i + 0.75) / 12.5);
    double dp = 1.0;
    for (int it = 0; it < 64; ++it) {
      double p0 = 1.0, p1 = x;
      for (int k = 2; k <= 12; ++k) {
        double pk = ((2.0 * k - 1.0) * x * p1 - (k - 1.0) * p0) / k;
        p0 = p1; p1 = pk;
      }
      dp = 12.0 * (x * p1 - p0) / (x * x - 1.0);
      x -= p1 / dp;
    }
    ct[i] = x; wt[i] = 2.0 / ((1.0 - x * x) * dp * dp);
  }
  int l1 = PL1[p], l2 = PL2[p], l3 = PL3[p];
  int d1 = 2 * l1 + 1, d2 = 2 * l2 + 1, d3 = 2 * l3 + 1;
  double acc[125];
  for (int i = 0; i < d1 * d2 * d3; ++i) acc[i] = 0.0;
  for (int it = 0; it < 12; ++it) {
    double z = ct[it];
    double st = sqrt(fmax(0.0, 1.0 - z * z));
    for (int ip = 0; ip < 24; ++ip) {
      double phi = 2.0 * PI * (ip + 0.5) / 24.0;
      double x = st * cos(phi), y = st * sin(phi);
      double W = wt[it] * (2.0 * PI / 24.0);
      double Ya[5], Yb[5], Yc[5];
      sh_ortho_d(x, y, z, l1, Ya);
      sh_ortho_d(x, y, z, l2, Yb);
      sh_ortho_d(x, y, z, l3, Yc);
      for (int i = 0; i < d1; ++i)
        for (int j = 0; j < d2; ++j)
          for (int k = 0; k < d3; ++k)
            acc[(i * d2 + j) * d3 + k] += W * Ya[i] * Yb[j] * Yc[k];
    }
  }
  double nrm = 0.0;
  for (int i = 0; i < d1 * d2 * d3; ++i) nrm += acc[i] * acc[i];
  nrm = sqrt(nrm);
  for (int i = 0; i < d1 * d2 * d3; ++i) cg[CGOFF[p] + i] = (float)(acc[i] / nrm);
}

// ---------------------------------------------------------------------------
// Node features h0 = [emb_w[atoms] | t], zero-padded to HID=288.
// ---------------------------------------------------------------------------
__global__ void build_h0_kernel(const float* __restrict__ x, const int* __restrict__ atoms,
                                const float* __restrict__ emb_w, float* __restrict__ h) {
  int tid = blockIdx.x * blockDim.x + threadIdx.x;
  if (tid >= N_NODES * HID) return;
  int n = tid / HID, c = tid % HID;
  float v = 0.f;
  if (c < EMB)            v = emb_w[atoms[n] * EMB + c];
  else if (c < 2 * EMB)   v = x[(n / APM) * XSTR + 3 * APM + (c - EMB)];
  h[tid] = v;
}

// ---------------------------------------------------------------------------
// Per-edge geometry: spherical harmonics (reference sph_harm scaling) + radial
// Bessel basis with polynomial cutoff.
// ---------------------------------------------------------------------------
__global__ void edge_geom_kernel(const float* __restrict__ x, const int* __restrict__ eidx,
                                 float* __restrict__ sh, float* __restrict__ ef) {
  int e = blockIdx.x * blockDim.x + threadIdx.x;
  if (e >= N_EDGES) return;
  int s = eidx[e], d = eidx[N_EDGES + e];
  float ax = x[(s / APM) * XSTR + (s % APM) * 3 + 0] - x[(d / APM) * XSTR + (d % APM) * 3 + 0];
  float ay = x[(s / APM) * XSTR + (s % APM) * 3 + 1] - x[(d / APM) * XSTR + (d % APM) * 3 + 1];
  float az = x[(s / APM) * XSTR + (s % APM) * 3 + 2] - x[(d / APM) * XSTR + (d % APM) * 3 + 2];
  float r = sqrtf(ax * ax + ay * ay + az * az);
  float inv = 1.f / r;
  float vx = ax * inv, vy = ay * inv, vz = az * inv;
  const float s3 = 1.7320508075688772f;   // sqrt(3)
  const float s15 = 3.872983346207417f;   // sqrt(15)
  const float h5 = 1.118033988749895f;    // 0.5*sqrt(5)
  const float h15 = 1.9364916731037085f;  // 0.5*sqrt(15)
  float* o = sh + (size_t)e * 9;
  o[0] = 1.f;
  o[1] = s3 * vy; o[2] = s3 * vz; o[3] = s3 * vx;
  o[4] = s15 * vx * vy; o[5] = s15 * vy * vz;
  o[6] = h5 * (3.f * vz * vz - 1.f);
  o[7] = s15 * vx * vz; o[8] = h15 * (vx * vx - vy * vy);
  // radial
  float u = r * 0.1f;                     // r / R_MAX
  float u5 = u * u * u * u * u;
  float f = 1.f - 21.f * u5 + 35.f * u5 * u - 15.f * u5 * u * u;
  if (u >= 1.f) f = 0.f;
  const float pref = 0.4472135954999579f; // sqrt(2/10)
  const float PIf = 3.14159265358979323846f;
  for (int k = 0; k < NB; ++k)
    ef[(size_t)e * NB + k] = pref * sinf((k + 1) * PIf * r * 0.1f) * inv * f;
}

// ---------------------------------------------------------------------------
// Stage 1 of radial MLP: H1 = relu(ef @ fc1 + b1), written directly as bf16 in
// CDNA5 WMMA A-fragment order (16-bit A 16x32 layout, ISA 7.12.2).
// ---------------------------------------------------------------------------
__global__ void mlp_pack_kernel(const float* __restrict__ ef, const float* __restrict__ fc1w,
                                const float* __restrict__ fc1b, unsigned short* __restrict__ Apack) {
  int tid = blockIdx.x * blockDim.x + threadIdx.x; // N_EDGES*MLP
  int e = tid >> 8, m = tid & 255;
  float v = fc1b[m];
#pragma unroll
  for (int k = 0; k < NB; ++k) v += ef[e * NB + k] * fc1w[k * MLP + m];
  v = fmaxf(v, 0.f);
  int et = e >> 4, m16 = e & 15, ks = m >> 5, kk = m & 31;
  int lane = m16 + (((kk >> 3) & 1) << 4);
  int j = ((kk >> 4) << 3) | (kk & 7);
  Apack[((((size_t)et * 8 + ks) * 32 + lane) << 4) + j] = f2bf(v);
}

// fc2 (MLP x NUM) -> bf16 B-fragment order (16-bit B 32x16 layout).
__global__ void pack_b_kernel(const float* __restrict__ fc2w, unsigned short* __restrict__ Bpack,
                              int NUMc) {
  int tid = blockIdx.x * blockDim.x + threadIdx.x;
  if (tid >= MLP * NUMc) return;
  int c = tid / NUMc, n = tid % NUMc;
  int nt = n >> 4, ks = c >> 5, kk = c & 31;
  int lane = (n & 15) + (((kk >> 3) & 1) << 4);
  int j = ((kk >> 4) << 3) | (kk & 7);
  Bpack[((((size_t)nt * 8 + ks) * 32 + lane) << 4) + j] = f2bf(fc2w[tid]);
}

// ---------------------------------------------------------------------------
// WMMA GEMM: Wc[e,n] = H1[e,:] @ fc2[:,n] + bias[n] for a CHUNK of edges.
// One wave per 16x16 tile, K=256 -> 8x v_wmma_f32_16x16x32_bf16.
// ---------------------------------------------------------------------------
__global__ void gemm_wmma_kernel(const unsigned short* __restrict__ Apack,
                                 const unsigned short* __restrict__ Bpack,
                                 const float* __restrict__ bias,
                                 float* __restrict__ Wc, int c0, int NUMc) {
  int lane = threadIdx.x & 31;
  int wave = blockIdx.x * (blockDim.x >> 5) + (threadIdx.x >> 5);
  int ntn = NUMc >> 4;
  int tiles = (CHUNK >> 4) * ntn;
  if (wave >= tiles) return;                 // wave-uniform; EXEC stays all-ones
  int mt = wave / ntn, nt = wave % ntn;
  int et = (c0 >> 4) + mt;
  const v16bf* A = reinterpret_cast<const v16bf*>(Apack) + (size_t)et * 8 * 32 + lane;
  const v16bf* B = reinterpret_cast<const v16bf*>(Bpack) + (size_t)nt * 8 * 32 + lane;
  v8f acc = {0.f, 0.f, 0.f, 0.f, 0.f, 0.f, 0.f, 0.f};
#pragma unroll
  for (int ks = 0; ks < 8; ++ks) {
    if (ks < 7) {                            // pull next fragments toward L0/L2
      __builtin_prefetch((const void*)(A + (ks + 1) * 32), 0, 3);
      __builtin_prefetch((const void*)(B + (ks + 1) * 32), 0, 3);
    }
    v16bf a = A[ks * 32];
    v16bf b = B[ks * 32];
    acc = __builtin_amdgcn_wmma_f32_16x16x32_bf16(false, a, false, b, (short)0, acc, false, false);
  }
  int n = (nt << 4) + (lane & 15);
  float bv = bias[n];
  int mrow = (mt << 4) + ((lane >> 4) << 3);  // D layout: m = (lane/16)*8 + r
#pragma unroll
  for (int r = 0; r < 8; ++r)
    Wc[(size_t)(mrow + r) * NUMc + n] = acc[r] + bv;
}

// ---------------------------------------------------------------------------
// Layer-0 conv: paths (0,l2,l2), mul_in=64, scalar source features.
// One wave per edge (8 waves/block). Lane = output channel v.
// ---------------------------------------------------------------------------
__global__ void conv0_kernel(const float* __restrict__ Wc, const float* __restrict__ h,
                             const float* __restrict__ sh, const int* __restrict__ eidx,
                             const float* __restrict__ cg, float* __restrict__ feats, int c0) {
  __shared__ float cgs[CGTOT];
  __shared__ float hd[8][64];
  int lane = threadIdx.x & 31, wv = threadIdx.x >> 5;
  for (int i = threadIdx.x; i < CGTOT; i += blockDim.x) cgs[i] = cg[i];
  int e = c0 + blockIdx.x * 8 + wv;
  int s = eidx[e], d = eidx[N_EDGES + e];
  hd[wv][lane]      = h[(size_t)d * HID + lane];
  hd[wv][32 + lane] = h[(size_t)d * HID + 32 + lane];
  float shv[9];
#pragma unroll
  for (int j = 0; j < 9; ++j) shv[j] = sh[(size_t)e * 9 + j];
  __syncthreads();
  const float* Wrow = Wc + (size_t)(e - c0) * NUM0;
  float m0 = 0.f, m1[3] = {0, 0, 0}, m2[5] = {0, 0, 0, 0, 0};
#pragma unroll
  for (int p = 0; p < 3; ++p) {
    const int l2 = p, d3 = 2 * l2 + 1;
    float S[5];
#pragma unroll
    for (int k = 0; k < d3; ++k) {
      float a = 0.f;
#pragma unroll
      for (int j = 0; j < d3; ++j) a += shv[SHOFF[l2] + j] * cgs[CGOFF[p] + j * d3 + k];
      S[k] = a;
    }
    float g = 0.f;
    for (int u = 0; u < 64; ++u) g += Wrow[p * 2048 + u * 32 + lane] * hd[wv][u];
    if (l2 == 0) m0 += g * S[0];
    else if (l2 == 1) { for (int k = 0; k < 3; ++k) m1[k] += g * S[k]; }
    else              { for (int k = 0; k < 5; ++k) m2[k] += g * S[k]; }
  }
  const float scl = 0.125f;                  // 1/sqrt(64*1)
  float* fs = feats + (size_t)s * HID;
  atomicAdd(&fs[lane], m0 * scl);
  for (int k = 0; k < 3; ++k) atomicAdd(&fs[32 + lane * 3 + k], m1[k] * scl);
  for (int k = 0; k < 5; ++k) atomicAdd(&fs[128 + lane * 5 + k], m2[k] * scl);
}

// ---------------------------------------------------------------------------
// Layers 1-2 conv: all 11 paths, mul_in=32. Lane = channel u for t, channel v
// for the message; t exchanged through LDS (block-uniform barriers).
// ---------------------------------------------------------------------------
__global__ void conv12_kernel(const float* __restrict__ Wc, const float* __restrict__ h,
                              const float* __restrict__ sh, const int* __restrict__ eidx,
                              const float* __restrict__ cg, float* __restrict__ feats, int c0) {
  __shared__ float cgs[CGTOT];
  __shared__ float tl[8][32 * 5];
  int lane = threadIdx.x & 31, wv = threadIdx.x >> 5;
  for (int i = threadIdx.x; i < CGTOT; i += blockDim.x) cgs[i] = cg[i];
  int e = c0 + blockIdx.x * 8 + wv;
  int s = eidx[e], d = eidx[N_EDGES + e];
  const float* hd = h + (size_t)d * HID;
  float a0 = hd[lane];
  float a1[3], a2[5];
#pragma unroll
  for (int i = 0; i < 3; ++i) a1[i] = hd[32 + lane * 3 + i];
#pragma unroll
  for (int i = 0; i < 5; ++i) a2[i] = hd[128 + lane * 5 + i];
  float shv[9];
#pragma unroll
  for (int j = 0; j < 9; ++j) shv[j] = sh[(size_t)e * 9 + j];
  const float* Wrow = Wc + (size_t)(e - c0) * NUM1;
  float m0 = 0.f, m1[3] = {0, 0, 0}, m2[5] = {0, 0, 0, 0, 0};
  __syncthreads();
#pragma unroll
  for (int p = 0; p < NPATH; ++p) {
    const int l1 = PL1[p], l2 = PL2[p], l3 = PL3[p];
    const int d1 = 2 * l1 + 1, d2 = 2 * l2 + 1, d3 = 2 * l3 + 1;
    // t[u=lane][k] = sum_{i,j} h_l1[u,i] * sh_l2[j] * CG[i,j,k]
#pragma unroll
    for (int k = 0; k < d3; ++k) {
      float acc = 0.f;
#pragma unroll
      for (int i = 0; i < d1; ++i) {
        float hv = (l1 == 0) ? a0 : (l1 == 1 ? a1[i] : a2[i]);
#pragma unroll
        for (int j = 0; j < d2; ++j)
          acc += hv * shv[SHOFF[l2] + j] * cgs[CGOFF[p] + (i * d2 + j) * d3 + k];
      }
      tl[wv][lane * d3 + k] = acc;
    }
    __syncthreads();
    // m[v=lane][k] = sum_u W[u,v] * t[u,k]
    const float* Wp = Wrow + p * EMB * EMB;
    float mk[5] = {0, 0, 0, 0, 0};
    for (int u = 0; u < 32; ++u) {
      float wval = Wp[u * 32 + lane];
#pragma unroll
      for (int k = 0; k < d3; ++k) mk[k] += wval * tl[wv][u * d3 + k];
    }
    if (l3 == 0) m0 += mk[0];
    else if (l3 == 1) { for (int k = 0; k < 3; ++k) m1[k] += mk[k]; }
    else              { for (int k = 0; k < 5; ++k) m2[k] += mk[k]; }
    __syncthreads();
  }
  const float s0 = 0.10206207261596575f;  // 1/sqrt(32*3)
  const float s1 = 0.08838834764831845f;  // 1/sqrt(32*4)
  float* fs = feats + (size_t)s * HID;
  atomicAdd(&fs[lane], m0 * s0);
  for (int k = 0; k < 3; ++k) atomicAdd(&fs[32 + lane * 3 + k], m1[k] * s1);
  for (int k = 0; k < 5; ++k) atomicAdd(&fs[128 + lane * 5 + k], m2[k] * s1);
}

// ---------------------------------------------------------------------------
// Batch statistics: per-channel mean/var of l=0, mean-square norms of l=1,2.
// ---------------------------------------------------------------------------
__global__ void bnstats_kernel(const float* __restrict__ feats, float* __restrict__ stats) {
  __shared__ float r0[256], r1[256], r2[256], r3[256];
  int c = blockIdx.x, t = threadIdx.x;
  float s = 0.f, s2 = 0.f, q1 = 0.f, q2 = 0.f;
  for (int n = t; n < N_NODES; n += 256) {
    const float* f = feats + (size_t)n * HID;
    float v = f[c]; s += v; s2 += v * v;
    for (int k = 0; k < 3; ++k) { float a = f[32 + c * 3 + k]; q1 += a * a; }
    for (int k = 0; k < 5; ++k) { float a = f[128 + c * 5 + k]; q2 += a * a; }
  }
  r0[t] = s; r1[t] = s2; r2[t] = q1; r3[t] = q2;
  __syncthreads();
  for (int off = 128; off > 0; off >>= 1) {
    if (t < off) { r0[t] += r0[t + off]; r1[t] += r1[t + off]; r2[t] += r2[t + off]; r3[t] += r3[t + off]; }
    __syncthreads();
  }
  if (t == 0) {
    float mean = r0[0] / N_NODES;
    stats[c]      = mean;
    stats[32 + c] = r1[0] / N_NODES - mean * mean;
    stats[64 + c] = r2[0] / (N_NODES * 3);
    stats[96 + c] = r3[0] / (N_NODES * 5);
  }
}

// ---------------------------------------------------------------------------
// Batchnorm + equivariant product block + linear mix + residual (one wave per
// node, lane = channel u; wlin mixes channels through LDS).
// ---------------------------------------------------------------------------
__global__ void product_kernel(const float* __restrict__ feats, const float* __restrict__ stats,
                               const float* __restrict__ hold, const float* __restrict__ cg,
                               const float* __restrict__ bnw, const float* __restrict__ bnb,
                               const float* __restrict__ w1, const float* __restrict__ w2,
                               const float* __restrict__ w3, const float* __restrict__ wlin,
                               float* __restrict__ hnew) {
  __shared__ float cgs[CGTOT];
  __shared__ float Bl[8][32][9];
  int lane = threadIdx.x & 31, wv = threadIdx.x >> 5;
  for (int i = threadIdx.x; i < CGTOT; i += blockDim.x) cgs[i] = cg[i];
  int n = blockIdx.x * 8 + wv;
  const float* f = feats + (size_t)n * HID;
  const float eps = 1e-5f;
  float A0, A1[3], A2[5];
  {
    float mean = stats[lane], var = stats[32 + lane];
    A0 = (f[lane] - mean) * rsqrtf(var + eps) * bnw[lane] + bnb[lane];
  }
  {
    float inv = rsqrtf(stats[64 + lane] + eps) * bnw[32 + lane];
    for (int k = 0; k < 3; ++k) A1[k] = f[32 + lane * 3 + k] * inv;
  }
  {
    float inv = rsqrtf(stats[96 + lane] + eps) * bnw[64 + lane];
    for (int k = 0; k < 5; ++k) A2[k] = f[128 + lane * 5 + k] * inv;
  }
  float B0 = A0 * w1[lane];
  float B1[3], B2[5];
  for (int k = 0; k < 3; ++k) B1[k] = A1[k] * w1[32 + lane];
  for (int k = 0; k < 5; ++k) B2[k] = A2[k] * w1[64 + lane];
  __syncthreads();
  float T[35];
#pragma unroll
  for (int p = 0; p < NPATH; ++p) {
    const int l1 = PL1[p], l2 = PL2[p], l3 = PL3[p];
    const int d1 = 2 * l1 + 1, d2 = 2 * l2 + 1, d3 = 2 * l3 + 1;
    float w2v = w2[p * EMB + lane];
#pragma unroll
    for (int k = 0; k < d3; ++k) {
      float acc = 0.f;
#pragma unroll
      for (int i = 0; i < d1; ++i) {
        float av = (l1 == 0) ? A0 : (l1 == 1 ? A1[i] : A2[i]);
#pragma unroll
        for (int j = 0; j < d2; ++j) {
          float bvv = (l2 == 0) ? A0 : (l2 == 1 ? A1[j] : A2[j]);
          acc += av * bvv * cgs[CGOFF[p] + (i * d2 + j) * d3 + k];
        }
      }
      T[TOFF[p] + k] = acc;
      if (l3 == 0) B0 += w2v * acc;
      else if (l3 == 1) B1[k] += w2v * acc;
      else              B2[k] += w2v * acc;
    }
  }
#pragma unroll
  for (int ci = 0; ci < NCOMBO; ++ci) {
    const int i = CBI[ci], j = CBJ[ci];
    const int L = PL3[i], lp = PL2[j], l3 = PL3[j];
    const int dk = 2 * L + 1, dj = 2 * lp + 1, dm = 2 * l3 + 1;
    float w3v = w3[ci * EMB + lane];
#pragma unroll
    for (int m = 0; m < dm; ++m) {
      float acc = 0.f;
#pragma unroll
      for (int k = 0; k < dk; ++k) {
        float tv = T[TOFF[i] + k];
#pragma unroll
        for (int jj = 0; jj < dj; ++jj) {
          float av = (lp == 0) ? A0 : (lp == 1 ? A1[jj] : A2[jj]);
          acc += tv * av * cgs[CGOFF[j] + (k * dj + jj) * dm + m];
        }
      }
      if (l3 == 0) B0 += w3v * acc;
      else if (l3 == 1) B1[m] += w3v * acc;
      else              B2[m] += w3v * acc;
    }
  }
  Bl[wv][lane][0] = B0;
  for (int k = 0; k < 3; ++k) Bl[wv][lane][1 + k] = B1[k];
  for (int k = 0; k < 5; ++k) Bl[wv][lane][4 + k] = B2[k];
  __syncthreads();
  const float inv = 0.17677669529663687f;  // 1/sqrt(EMB)
  float* hn = hnew + (size_t)n * HID;
  const float* ho = hold + (size_t)n * HID;
  {
    float acc = 0.f;
    for (int u = 0; u < 32; ++u) acc += Bl[wv][u][0] * wlin[u * 32 + lane];
    hn[lane] = acc * inv + ho[lane];
  }
  for (int k = 0; k < 3; ++k) {
    float acc = 0.f;
    for (int u = 0; u < 32; ++u) acc += Bl[wv][u][1 + k] * wlin[1024 + u * 32 + lane];
    hn[32 + lane * 3 + k] = acc * inv + ho[32 + lane * 3 + k];
  }
  for (int k = 0; k < 5; ++k) {
    float acc = 0.f;
    for (int u = 0; u < 32; ++u) acc += Bl[wv][u][4 + k] * wlin[2048 + u * 32 + lane];
    hn[128 + lane * 5 + k] = acc * inv + ho[128 + lane * 5 + k];
  }
}

// ---------------------------------------------------------------------------
// Pool over atoms + 2-layer prediction head. One wave per molecule.
// ---------------------------------------------------------------------------
__global__ void predict_kernel(const float* __restrict__ h, const float* __restrict__ pw1,
                               const float* __restrict__ pb1, const float* __restrict__ pw2,
                               const float* __restrict__ pb2, float* __restrict__ out) {
  __shared__ float pooled[32];
  __shared__ float ov[32];
  int b = blockIdx.x, t = threadIdx.x;
  float s = 0.f;
  for (int a = 0; a < APM; ++a) s += h[(size_t)(b * APM + a) * HID + t];
  pooled[t] = s;
  __syncthreads();
  float o = pb1[t];
  for (int i = 0; i < 32; ++i) o += pooled[i] * pw1[i * 32 + t];
  ov[t] = fmaxf(o, 0.f) * pw2[t];
  __syncthreads();
  if (t == 0) {
    float acc = pb2[0];
    for (int i = 0; i < 32; ++i) acc += ov[i];
    out[b] = acc;
  }
}

// ---------------------------------------------------------------------------
extern "C" void kernel_launch(void* const* d_in, const int* in_sizes, int n_in,
                              void* d_out, int out_size, void* d_ws, size_t ws_size,
                              hipStream_t stream) {
  (void)in_sizes; (void)n_in; (void)out_size; (void)ws_size;
  const float* x      = (const float*)d_in[0];
  const int*   atoms  = (const int*)d_in[1];
  const int*   eidx   = (const int*)d_in[2];
  const float* emb_w  = (const float*)d_in[3];
  const float* fc1_w0 = (const float*)d_in[4];
  const float* fc1_b0 = (const float*)d_in[5];
  const float* fc2_w0 = (const float*)d_in[6];
  const float* fc2_b0 = (const float*)d_in[7];
  const float* fc1_w  = (const float*)d_in[8];
  const float* fc1_b  = (const float*)d_in[9];
  const float* fc2_w  = (const float*)d_in[10];
  const float* fc2_b  = (const float*)d_in[11];
  const float* bn_w   = (const float*)d_in[12];
  const float* bn_b   = (const float*)d_in[13];
  const float* w1     = (const float*)d_in[14];
  const float* w2     = (const float*)d_in[15];
  const float* w3     = (const float*)d_in[16];
  const float* wlin   = (const float*)d_in[17];
  const float* pw1    = (const float*)d_in[18];
  const float* pb1    = (const float*)d_in[19];
  const float* pw2    = (const float*)d_in[20];
  const float* pb2    = (const float*)d_in[21];

  // ---- scratch layout (~48 MB total) ----
  char* ws = (char*)d_ws;
  size_t off = 0;
  auto alloc = [&](size_t bytes) -> char* {
    char* p = ws + off; off += (bytes + 255) & ~size_t(255); return p;
  };
  float* cg    = (float*)alloc(CGTOT * 4);
  float* hA    = (float*)alloc((size_t)N_NODES * HID * 4);
  float* hB    = (float*)alloc((size_t)N_NODES * HID * 4);
  float* feats = (float*)alloc((size_t)N_NODES * HID * 4);
  float* shb   = (float*)alloc((size_t)N_EDGES * 9 * 4);
  float* efb   = (float*)alloc((size_t)N_EDGES * NB * 4);
  unsigned short* Apack = (unsigned short*)alloc((size_t)N_EDGES * MLP * 2);
  unsigned short* Bpack = (unsigned short*)alloc((size_t)MLP * NUM1 * 2);
  float* Wc    = (float*)alloc((size_t)CHUNK * NUM1 * 4);
  float* stats = (float*)alloc(128 * 4);

  compute_cg_kernel<<<1, 32, 0, stream>>>(cg);
  build_h0_kernel<<<(N_NODES * HID + 255) / 256, 256, 0, stream>>>(x, atoms, emb_w, hA);
  edge_geom_kernel<<<(N_EDGES + 255) / 256, 256, 0, stream>>>(x, eidx, shb, efb);

  float* hcur = hA;
  float* hnext = hB;
  for (int li = 0; li < 3; ++li) {
    const float* f1w = (li == 0) ? fc1_w0 : fc1_w + (size_t)(li - 1) * NB * MLP;
    const float* f1b = (li == 0) ? fc1_b0 : fc1_b + (size_t)(li - 1) * MLP;
    const float* f2w = (li == 0) ? fc2_w0 : fc2_w + (size_t)(li - 1) * MLP * NUM1;
    const float* f2b = (li == 0) ? fc2_b0 : fc2_b + (size_t)(li - 1) * NUM1;
    const int NUMc = (li == 0) ? NUM0 : NUM1;

    mlp_pack_kernel<<<N_EDGES * MLP / 256, 256, 0, stream>>>(efb, f1w, f1b, Apack);
    pack_b_kernel<<<(MLP * NUMc + 255) / 256, 256, 0, stream>>>(f2w, Bpack, NUMc);
    hipMemsetAsync(feats, 0, (size_t)N_NODES * HID * 4, stream);

    const int tiles = (CHUNK / 16) * (NUMc / 16);
    const int gblocks = (tiles + 3) / 4;             // 4 waves / block
    for (int c0 = 0; c0 < N_EDGES; c0 += CHUNK) {
      gemm_wmma_kernel<<<gblocks, 128, 0, stream>>>(Apack, Bpack, f2b, Wc, c0, NUMc);
      if (li == 0)
        conv0_kernel<<<CHUNK / 8, 256, 0, stream>>>(Wc, hcur, shb, eidx, cg, feats, c0);
      else
        conv12_kernel<<<CHUNK / 8, 256, 0, stream>>>(Wc, hcur, shb, eidx, cg, feats, c0);
    }
    bnstats_kernel<<<EMB, 256, 0, stream>>>(feats, stats);
    product_kernel<<<N_NODES / 8, 256, 0, stream>>>(
        feats, stats, hcur, cg,
        bn_w + (size_t)li * 3 * EMB, bn_b + (size_t)li * EMB,
        w1 + (size_t)li * 3 * EMB, w2 + (size_t)li * NPATH * EMB,
        w3 + (size_t)li * NCOMBO * EMB, wlin + (size_t)li * 3 * EMB * EMB, hnext);
    float* tmp = hcur; hcur = hnext; hnext = tmp;
  }

  predict_kernel<<<BS, 32, 0, stream>>>(hcur, pw1, pb1, pw2, pb2, (float*)d_out);
}
